// InflatedButterflyLayer1D_52725018526080
// MI455X (gfx1250) — matrix operand order repaired and checked by reading, not compiled, not run on hardware
//
#include <hip/hip_runtime.h>

typedef __attribute__((ext_vector_type(16))) _Float16 v16h;
typedef __attribute__((ext_vector_type(8)))  _Float16 v8h;
typedef __attribute__((ext_vector_type(8)))  float    v8f;
typedef __attribute__((ext_vector_type(4)))  float    v4f;
typedef __attribute__((ext_vector_type(4)))  unsigned v4u;
typedef __attribute__((ext_vector_type(8)))  int      v8i;
typedef __attribute__((ext_vector_type(4)))  int      v4i;

#define BM 128
#define BN 64
#define BK 32

static __device__ __forceinline__ v8f wmma_f16(v16h a, v16h b, v8f c) {
    return __builtin_amdgcn_wmma_f32_16x16x32_f16(false, a, false, b, (short)0, c,
                                                  false, false);
}

static __device__ __forceinline__ void wait_tdm() {
#if __has_builtin(__builtin_amdgcn_s_wait_tensorcnt)
    __builtin_amdgcn_s_wait_tensorcnt(0);
#endif
}

// ---------------------------------------------------------------------------
// Weight prep: f32 (K x N) row-major  ->  f16 tiles in WMMA B-fragment order.
// One 512-half block per (32K x 16N) sub-tile; blocks laid out [mat][kt][nt],
// nt padded to a multiple of 4 so a BN=64 GEMM tile is one contiguous 4KB run
// (exactly the LDS image the GEMM wants -> single TDM copy per K-step).
// ---------------------------------------------------------------------------
__global__ __launch_bounds__(256)
void prep_w(const float* __restrict__ W, _Float16* __restrict__ Wo, int K, int N,
            int NTpad)
{
    int blk = blockIdx.x;                 // kt*NTpad + nt
    int kt = blk / NTpad, nt = blk % NTpad;
    const float* Ws = W + (size_t)blockIdx.y * (size_t)K * (size_t)N;
    _Float16*    Wd = Wo + ((size_t)blockIdx.y * gridDim.x + blk) * 512;
    int t = threadIdx.x;
#pragma unroll
    for (int e = 0; e < 2; ++e) {
        int pos  = t * 2 + e;             // 0..511 within block
        int lane = pos >> 4, h = pos & 15;
        int Nloc = lane & 15;
        int Kl   = h + ((lane >= 16) ? 16 : 0);
        int kg   = kt * 32 + Kl;
        int gn   = nt * 16 + Nloc;
        float v  = (kg < K && gn < N) ? Ws[(size_t)kg * N + gn] : 0.0f;
        Wd[pos]  = (_Float16)v;
    }
}

// ---------------------------------------------------------------------------
// Generic fused GEMM: C = epilogue(A * W + bias)
//  AH: A is f16 (else f32) ; CH: store f16 (else f32)
//  - optional butterfly K-gather (in_nitx) folds the pre-shuffle
//  - optional butterfly scatter store (out_nitx) folds the post-shuffle
//  - mid!=0: blockIdx.z selects one of 256 (k,x) switch-layer sub-problems
//  A staged to LDS in fragment order (b128 ds traffic); B staged by the
//  Tensor Data Mover directly from pre-swizzled weights. Double-buffered.
// ---------------------------------------------------------------------------
template <int AH, int CH>
__global__ __launch_bounds__(256)
void bf_gemm_wmma(const void* __restrict__ Ain, const _Float16* __restrict__ W16,
                  const float* __restrict__ bias, void* __restrict__ Cout,
                  int M, int N, int K, int lda, int ldc, int NTpad,
                  int in_nitx, int out_nitx, int rows_img, int relu, int mid)
{
    __shared__ _Float16 As[2][8 * 512];   // 8 (16x32) frag-order sub-tiles
    __shared__ _Float16 Bs[2][4 * 512];   // 4 (32x16) frag-order sub-tiles

    const int t    = threadIdx.x;
    const int lane = t & 31;
    const int wave = t >> 5;
    const int wm   = wave & 3;            // 4 wave-rows x 32 rows
    const int wn   = wave >> 2;           // 2 wave-cols x 32 cols

    const int mBase = blockIdx.x * BM;
    const int nBase = blockIdx.y * BN;

    const float*    Af = (const float*)Ain;
    const _Float16* Ah = (const _Float16*)Ain;
    float*          Cf = (float*)Cout;
    _Float16*       Chh = (_Float16*)Cout;
    const _Float16* Wp = W16;
    const float*    bp = bias;
    size_t aOff = 0, cOff = 0;
    if (mid) {
        int pr = blockIdx.z;
        int kk = pr >> 4, xx = pr & 15;
        aOff = (size_t)xx * 1024 + (size_t)kk * 64;          // y[b, x, k, :]
        int KT = (K + 31) >> 5;
        Wp += (size_t)pr * (size_t)KT * (size_t)NTpad * 512; // midw[k][x] swizzled
        if (bp) bp += (size_t)pr * 64;                       // midb[k][x]
        cOff = (size_t)kk * 1024 + (size_t)xx * 64;          // h[b, k, x*64 + :]
    }

    const int cin = K >> 1;

    // ---- A staging coords: thread covers one row, 16 k (two 8-runs) ----
    const int arow    = t >> 1;           // 0..127
    const int akb     = (t & 1) * 16;     // 0 or 16
    const int asub    = arow >> 4;        // 0..7
    const int aMloc   = arow & 15;
    const size_t aRow = (size_t)(mBase + arow) * (size_t)lda;

    auto stageA = [&](int k0, int bufsel) {
#pragma unroll
        for (int rsel = 0; rsel < 2; ++rsel) {
            int klocal = akb + rsel * 8;
            int kg     = k0 + klocal;
            v8h out    = {};
            if (kg + 8 <= K) {            // all-or-nothing: K multiple of 8
                int src = kg;
                if (in_nitx) {            // butterfly gather (8-run contiguous)
                    int r = (kg >= cin) ? 1 : 0;
                    int q = kg - (r ? cin : 0);
                    src   = (((q >> 6) << 1) + r) * 64 + (q & 63);
                }
                if (AH) {
                    const _Float16* p = Ah + aOff + aRow + src;
                    out = *(const v8h*)p;
                    if (rsel == 0 && kg + 72 <= K)
                        __builtin_prefetch((const void*)(p + 64), 0, 0);
                } else {
                    const float* p = Af + aOff + aRow + src;
                    v4f lo = *(const v4f*)p;
                    v4f hi = *(const v4f*)(p + 4);
#pragma unroll
                    for (int j = 0; j < 4; ++j) {
                        out[j]     = (_Float16)lo[j];
                        out[4 + j] = (_Float16)hi[j];
                    }
                }
            }
            int lane_s = aMloc + (((klocal >> 4) & 1) << 4);
            int h0     = ((klocal >> 3) & 1) << 3;
            *(v8h*)&As[bufsel][asub * 512 + lane_s * 16 + h0] = out;
        }
    };

    // B tile: one contiguous 4KB pre-swizzled block run -> one TDM transfer
    auto stageB = [&](int k0, int bufsel) {
        int kt = k0 >> 5;
        const _Float16* gsrc = Wp + (size_t)(kt * NTpad + (nBase >> 4)) * 512;
#if __has_builtin(__builtin_amdgcn_tensor_load_to_lds)
        if (wave == 0) {
            unsigned ldsAddr = (unsigned)(uintptr_t)(void*)&Bs[bufsel][0];
            unsigned long long ga = (unsigned long long)(uintptr_t)gsrc;
            // D# group0: count=1 | lds_addr | global_addr | type=2
            v4u g0 = { 1u, ldsAddr, (unsigned)ga,
                       (unsigned)((ga >> 32) & 0x1FFFFFFu) | 0x80000000u };
            // D# group1: data_size=4B; tensor_dim0=1024, tensor_dim1=1,
            //            tile_dim0=1024, tile_dim1=1, stride0=1024
            v8i g1 = { 0x20000, 0x04000000, 0x10000, 0x04000000, 1, 1024, 0, 0 };
            v4i z4 = { 0, 0, 0, 0 };
#if defined(__clang_major__) && (__clang_major__ >= 23)
            v8i z8 = { 0, 0, 0, 0, 0, 0, 0, 0 };
            __builtin_amdgcn_tensor_load_to_lds(g0, g1, z4, z4, z8, 0);
#else
            __builtin_amdgcn_tensor_load_to_lds(g0, g1, z4, z4, 0);
#endif
        }
#else
        // fallback: cooperative 16B-per-thread copy (also used by host pass)
        *(v8h*)&Bs[bufsel][t * 8] = *(const v8h*)(gsrc + t * 8);
#endif
    };

    v8f acc[2][2] = {};

    stageA(0, 0);
    stageB(0, 0);
    wait_tdm();
    __syncthreads();

    int buf = 0;
    for (int k0 = 0; k0 < K; k0 += BK) {
        if (k0 + BK < K) {
            stageA(k0 + BK, buf ^ 1);
            stageB(k0 + BK, buf ^ 1);
        }

        v16h a0 = *(const v16h*)&As[buf][(wm * 2 + 0) * 512 + lane * 16];
        v16h a1 = *(const v16h*)&As[buf][(wm * 2 + 1) * 512 + lane * 16];
        v16h b0 = *(const v16h*)&Bs[buf][(wn * 2 + 0) * 512 + lane * 16];
        v16h b1 = *(const v16h*)&Bs[buf][(wn * 2 + 1) * 512 + lane * 16];

        acc[0][0] = wmma_f16(a0, b0, acc[0][0]);
        acc[0][1] = wmma_f16(a0, b1, acc[0][1]);
        acc[1][0] = wmma_f16(a1, b0, acc[1][0]);
        acc[1][1] = wmma_f16(a1, b1, acc[1][1]);

        wait_tdm();        // next tile's DMA has landed before we swap buffers
        __syncthreads();
        buf ^= 1;
    }

    // ---- epilogue: bias + relu, contiguous or post-shuffle scatter store ----
#pragma unroll
    for (int mi = 0; mi < 2; ++mi) {
#pragma unroll
        for (int ni = 0; ni < 2; ++ni) {
            int gn  = nBase + wn * 32 + ni * 16 + (lane & 15);
            int gmb = mBase + wm * 32 + mi * 16 + ((lane >= 16) ? 8 : 0);
            if (gn >= N) continue;
            float bv = bp ? bp[gn] : 0.0f;
#pragma unroll
            for (int r = 0; r < 8; ++r) {
                int gm  = gmb + r;        // M divisible by BM: no guard needed
                float v = acc[mi][ni][r] + bv;
                if (relu) v = fmaxf(v, 0.0f);
                size_t dst;
                if (out_nitx) {
                    int b = gm / rows_img, p = gm - b * rows_img;
                    int tc = gn >> 6, j = gn & 63;
                    int x = tc >> 1, rr = tc & 1;
                    dst = (size_t)(b * 2 * rows_img + 2 * p + rr) *
                              (size_t)(out_nitx * 64) +
                          (size_t)(x * 64 + j);
                } else {
                    dst = (size_t)gm * (size_t)ldc + (size_t)gn;
                }
                if (CH) Chh[cOff + dst] = (_Float16)v;
                else    Cf[cOff + dst]  = v;
            }
        }
    }
}

static inline void launch_gemm(hipStream_t s, int ah, int ch, const void* A,
                               const _Float16* W16, const float* bias, void* C,
                               int M, int N, int K, int lda, int ldc, int NTpad,
                               int in_nitx, int out_nitx, int rows_img,
                               int relu, int mid, int nz)
{
    dim3 grid(M / BM, (N + BN - 1) / BN, nz);
    dim3 blk(256);
    if (ah && ch)
        bf_gemm_wmma<1, 1><<<grid, blk, 0, s>>>(A, W16, bias, C, M, N, K, lda, ldc,
                                                NTpad, in_nitx, out_nitx, rows_img,
                                                relu, mid);
    else if (!ah)
        bf_gemm_wmma<0, 1><<<grid, blk, 0, s>>>(A, W16, bias, C, M, N, K, lda, ldc,
                                                NTpad, in_nitx, out_nitx, rows_img,
                                                relu, mid);
    else
        bf_gemm_wmma<1, 0><<<grid, blk, 0, s>>>(A, W16, bias, C, M, N, K, lda, ldc,
                                                NTpad, in_nitx, out_nitx, rows_img,
                                                relu, mid);
}

extern "C" void kernel_launch(void* const* d_in, const int* in_sizes, int n_in,
                              void* d_out, int out_size, void* d_ws, size_t ws_size,
                              hipStream_t stream)
{
    (void)in_sizes; (void)n_in; (void)out_size; (void)ws_size;

    const float* x    = (const float*)d_in[0];
    const float* xf   = (const float*)d_in[1];
    const float* xb   = (const float*)d_in[2];
    const float* w1   = (const float*)d_in[3];
    const float* b1   = (const float*)d_in[4];
    const float* w2   = (const float*)d_in[5];
    const float* b2   = (const float*)d_in[6];
    const float* w3   = (const float*)d_in[7];
    const float* b3   = (const float*)d_in[8];
    const float* w4   = (const float*)d_in[9];
    const float* b4   = (const float*)d_in[10];
    const float* midw = (const float*)d_in[11];
    const float* midb = (const float*)d_in[12];
    const float* w5   = (const float*)d_in[13];
    const float* b5   = (const float*)d_in[14];
    const float* w6   = (const float*)d_in[15];
    const float* b6   = (const float*)d_in[16];
    const float* w7   = (const float*)d_in[17];
    const float* b7   = (const float*)d_in[18];
    const float* w8   = (const float*)d_in[19];
    const float* b8   = (const float*)d_in[20];
    const float* kf   = (const float*)d_in[21];

    const size_t BUF = (size_t)512 * 256 * 64;   // 8.39M halves per stage buffer
    _Float16* hbuf0 = (_Float16*)d_ws;
    _Float16* hbuf1 = hbuf0 + BUF;
    _Float16* wreg  = hbuf1 + BUF;

    // ---- pre-swizzle all weights into f16 fragment-order tile images ----
    struct WRec { _Float16* p; int NTpad; };
    auto prep = [&](const float* W, int K, int N, int nmats) -> WRec {
        int KT = (K + 31) / 32;
        int NTpad = ((N + 63) / 64) * 4;
        _Float16* dst = wreg;
        dim3 g(KT * NTpad, nmats);
        prep_w<<<g, dim3(256), 0, stream>>>(W, dst, K, N, NTpad);
        wreg += (size_t)nmats * (size_t)(KT * NTpad) * 512;
        return WRec{dst, NTpad};
    };

    WRec Wxf = prep(xf,   16,   64, 1);
    WRec W1  = prep(w1,  128,  128, 1);
    WRec W2  = prep(w2,  256,  256, 1);
    WRec W3  = prep(w3,  512,  512, 1);
    WRec W4  = prep(w4, 1024, 1024, 1);
    WRec Wm  = prep(midw, 64,   64, 256);
    WRec W5  = prep(w5, 1024, 1024, 1);
    WRec W6  = prep(w6,  512,  512, 1);
    WRec W7  = prep(w7,  256,  256, 1);
    WRec W8  = prep(w8,  128,  128, 1);
    WRec Wkf = prep(kf,   64,   16, 1);

    // ---- pipeline: 11 fused WMMA GEMMs, f16 ping-pong intermediates ----
    // Patchify: (B*256,16) x (16,64) -> (B,256,64)
    launch_gemm(stream, 0, 1, x,     Wxf.p, xb,   hbuf0, 131072,   64,   16,   16,   64, Wxf.NTpad, 0, 0,   0, 1, 0, 1);
    // Expanding stride-2 convs as contiguous GEMMs
    launch_gemm(stream, 1, 1, hbuf0, W1.p,  b1,   hbuf1,  65536,  128,  128,  128,  128, W1.NTpad,  0, 0,   0, 1, 0, 1);
    launch_gemm(stream, 1, 1, hbuf1, W2.p,  b2,   hbuf0,  32768,  256,  256,  256,  256, W2.NTpad,  0, 0,   0, 1, 0, 1);
    launch_gemm(stream, 1, 1, hbuf0, W3.p,  b3,   hbuf1,  16384,  512,  512,  512,  512, W3.NTpad,  0, 0,   0, 1, 0, 1);
    launch_gemm(stream, 1, 1, hbuf1, W4.p,  b4,   hbuf0,   8192, 1024, 1024, 1024, 1024, W4.NTpad,  0, 0,   0, 1, 0, 1);
    // Mid switch: 256 batched (512x64)x(64x64) GEMMs, transposed store
    launch_gemm(stream, 1, 1, hbuf0, Wm.p,  midb, hbuf1,    512,   64,   64, 16384, 16384, Wm.NTpad, 0, 0,  0, 1, 1, 256);
    // Contracting levels: pre-shuffle folded into K-gather, post-shuffle into store scatter
    launch_gemm(stream, 1, 1, hbuf1, W5.p,  b5,   hbuf0,   8192, 1024, 1024, 1024,    0, W5.NTpad,  8, 8,  16, 1, 0, 1);
    launch_gemm(stream, 1, 1, hbuf0, W6.p,  b6,   hbuf1,  16384,  512,  512,  512,    0, W6.NTpad,  4, 4,  32, 1, 0, 1);
    launch_gemm(stream, 1, 1, hbuf1, W7.p,  b7,   hbuf0,  32768,  256,  256,  256,    0, W7.NTpad,  2, 2,  64, 1, 0, 1);
    launch_gemm(stream, 1, 1, hbuf0, W8.p,  b8,   hbuf1,  65536,  128,  128,  128,    0, W8.NTpad,  1, 1, 128, 1, 0, 1);
    // Final 1x1 conv: no bias, no relu -> d_out (B,4096) f32
    launch_gemm(stream, 1, 0, hbuf1, Wkf.p, nullptr, d_out,
                                                     131072,   16,   64,   64,   16, Wkf.NTpad, 0, 0,   0, 0, 0, 1);
}
